// TripletHashLoss_86732569576123
// MI455X (gfx1250) — matrix operand order repaired and checked by reading, not compiled, not run on hardware
//
#include <hip/hip_runtime.h>
#include <hip/hip_bf16.h>

typedef __attribute__((ext_vector_type(16))) _Float16 v16h;
typedef __attribute__((ext_vector_type(8)))  float    v8f;

#define GRAM_ROWS 16

// ---------------------------------------------------------------------------
// Kernel 1: gram[0:16, :] = X[0:16,:] @ X^T  via v_wmma_f32_16x16x32_f16.
// One wave per 16x16 output tile; K = 128 fully unrolled in 4 steps of 32.
// Inputs are +/-1.0f so f16 conversion / f16 WMMA are bit-exact.
// Per-lane fragment data is contiguous in memory, so loads are float4
// (global_load_b128):
//   A (16x32 f16, ISA 7.12.2): lane m=lane&15 reads
//       arow[kk + half*8 + 0..7]  -> a[0..7]
//       arow[kk + 16 + half*8 + 0..7] -> a[8..15]
//   B (32x16): lane n=lane&15 reads brow[kk + half*8 + 0..7] -> b[0..7],
//       brow[kk + 16 + half*8 + 0..7] -> b[8..15]
//   C/D: lane n=lane&15, VGPR r -> M = r + half*8.
// ---------------------------------------------------------------------------
__global__ void __launch_bounds__(32)
gram_slab_wmma(const float* __restrict__ X, float* __restrict__ gram,
               int Bn, int L) {
    const int lane = threadIdx.x;      // 0..31
    const int idx  = lane & 15;
    const int half = lane >> 4;        // 0 or 1
    const int n0   = blockIdx.x * 16;

    const float* __restrict__ arow = X + (size_t)idx * L;          // A row m=idx
    const float* __restrict__ brow = X + (size_t)(n0 + idx) * L;   // B col n=idx

    v8f c = {};
#pragma unroll
    for (int kk = 0; kk < 128; kk += 32) {
        const float4 fa0 = *(const float4*)(arow + kk + half * 8);
        const float4 fa1 = *(const float4*)(arow + kk + half * 8 + 4);
        const float4 fa2 = *(const float4*)(arow + kk + 16 + half * 8);
        const float4 fa3 = *(const float4*)(arow + kk + 16 + half * 8 + 4);
        const float4 fb0 = *(const float4*)(brow + kk + half * 8);
        const float4 fb1 = *(const float4*)(brow + kk + half * 8 + 4);
        const float4 fb2 = *(const float4*)(brow + kk + 16 + half * 8);
        const float4 fb3 = *(const float4*)(brow + kk + 16 + half * 8 + 4);

        v16h a, b;
        a[0]  = (_Float16)fa0.x;  a[1]  = (_Float16)fa0.y;
        a[2]  = (_Float16)fa0.z;  a[3]  = (_Float16)fa0.w;
        a[4]  = (_Float16)fa1.x;  a[5]  = (_Float16)fa1.y;
        a[6]  = (_Float16)fa1.z;  a[7]  = (_Float16)fa1.w;
        a[8]  = (_Float16)fa2.x;  a[9]  = (_Float16)fa2.y;
        a[10] = (_Float16)fa2.z;  a[11] = (_Float16)fa2.w;
        a[12] = (_Float16)fa3.x;  a[13] = (_Float16)fa3.y;
        a[14] = (_Float16)fa3.z;  a[15] = (_Float16)fa3.w;

        b[0]  = (_Float16)fb0.x;  b[1]  = (_Float16)fb0.y;
        b[2]  = (_Float16)fb0.z;  b[3]  = (_Float16)fb0.w;
        b[4]  = (_Float16)fb1.x;  b[5]  = (_Float16)fb1.y;
        b[6]  = (_Float16)fb1.z;  b[7]  = (_Float16)fb1.w;
        b[8]  = (_Float16)fb2.x;  b[9]  = (_Float16)fb2.y;
        b[10] = (_Float16)fb2.z;  b[11] = (_Float16)fb2.w;
        b[12] = (_Float16)fb3.x;  b[13] = (_Float16)fb3.y;
        b[14] = (_Float16)fb3.z;  b[15] = (_Float16)fb3.w;

        c = __builtin_amdgcn_wmma_f32_16x16x32_f16(
                /*neg_a=*/false, a, /*neg_b=*/false, b,
                /*c_mod=*/(short)0, c, /*reuse_a=*/false, /*reuse_b=*/false);
    }
#pragma unroll
    for (int r = 0; r < 8; ++r) {
        const int m = r + half * 8;
        gram[(size_t)m * Bn + n0 + idx] = c[r];
    }
}

// ---------------------------------------------------------------------------
// Kernel 2: pos_loss = sum(max(margin - dot(X[q],X[d]), 0)^2). Wave per pair.
// ---------------------------------------------------------------------------
__global__ void __launch_bounds__(256)
pos_pairs(const float* __restrict__ X, const long long* __restrict__ pos,
          float* __restrict__ acc, int P, int L, float margin) {
    const int lane = threadIdx.x & 31;
    const int p = blockIdx.x * 8 + (threadIdx.x >> 5);
    if (p >= P) return;
    const int q = (int)pos[2 * p];
    const int d = (int)pos[2 * p + 1];
    const float* xq = X + (size_t)q * L;
    const float* xd = X + (size_t)d * L;
    float dot = 0.0f;
    for (int i = lane; i < L; i += 32) dot += xq[i] * xd[i];
#pragma unroll
    for (int m = 16; m >= 1; m >>= 1) dot += __shfl_xor(dot, m, 32);
    if (lane == 0) {
        float h = fmaxf(margin - dot, 0.0f);
        atomicAdd(acc, h * h);
    }
}

// ---------------------------------------------------------------------------
// Kernel 3: mark positive-pair mask restricted to rows [0, GRAM_ROWS).
// ---------------------------------------------------------------------------
__global__ void __launch_bounds__(256)
mark_mask(const long long* __restrict__ pos, unsigned char* __restrict__ mask,
          int Bn, int P) {
    const int p = blockIdx.x * blockDim.x + threadIdx.x;
    if (p >= P) return;
    const int q = (int)pos[2 * p];
    const int d = (int)pos[2 * p + 1];
    if (q < GRAM_ROWS) mask[(size_t)q * Bn + d] = 1;
    if (d < GRAM_ROWS) mask[(size_t)d * Bn + q] = 1;
}

// ---------------------------------------------------------------------------
// Kernel 4: ordered selection of first P candidates (flat row-major) within
// the 16-row slab; neg_loss = sum over selected of max(g + margin, 0)^2.
// Single 1024-thread block; LDS Hillis-Steele scan of per-chunk counts.
// ---------------------------------------------------------------------------
__global__ void __launch_bounds__(1024)
neg_select(const float* __restrict__ gram, const unsigned char* __restrict__ mask,
           float* __restrict__ neg_acc, int Bn, int P, float margin) {
    __shared__ int   scnt[1024];
    __shared__ float ssum[1024];
    const int t = threadIdx.x;
    const int total = GRAM_ROWS * Bn;
    const int per = (total + 1023) / 1024;
    const int start = t * per;
    const int end = min(start + per, total);

    int cnt = 0;
    for (int i = start; i < end; ++i) {
        const int row = i / Bn, col = i - row * Bn;
        if (col != row && mask[i] == 0) ++cnt;
    }
    scnt[t] = cnt;
    __syncthreads();
    for (int off = 1; off < 1024; off <<= 1) {
        const int v = (t >= off) ? scnt[t - off] : 0;
        __syncthreads();
        scnt[t] += v;
        __syncthreads();
    }
    int rank = scnt[t] - cnt;   // exclusive prefix = global candidate rank

    float sum = 0.0f;
    if (rank < P) {
        for (int i = start; i < end && rank < P; ++i) {
            const int row = i / Bn, col = i - row * Bn;
            if (col != row && mask[i] == 0) {
                const float v = fmaxf(gram[i] + margin, 0.0f);
                sum += v * v;
                ++rank;
            }
        }
    }
    ssum[t] = sum;
    __syncthreads();
    for (int off = 512; off >= 1; off >>= 1) {
        if (t < off) ssum[t] += ssum[t + off];
        __syncthreads();
    }
    if (t == 0) *neg_acc = ssum[0];
}

// ---------------------------------------------------------------------------
// Kernel 5: quantization loss partial sums: sum |abs(c) - 1|.
// ---------------------------------------------------------------------------
__global__ void __launch_bounds__(256)
quant_sum(const float* __restrict__ C, float* __restrict__ acc, int n) {
    __shared__ float s[256];
    float local = 0.0f;
    for (int i = blockIdx.x * blockDim.x + threadIdx.x; i < n;
         i += gridDim.x * blockDim.x)
        local += fabsf(fabsf(C[i]) - 1.0f);
    s[threadIdx.x] = local;
    __syncthreads();
    for (int off = 128; off >= 1; off >>= 1) {
        if (threadIdx.x < off) s[threadIdx.x] += s[threadIdx.x + off];
        __syncthreads();
    }
    if (threadIdx.x == 0) atomicAdd(acc, s[0]);
}

// ---------------------------------------------------------------------------
// Kernel 6: per-column sums for bit-balance; thread = column, block = 128 rows.
// ---------------------------------------------------------------------------
__global__ void __launch_bounds__(128)
bit_colsum(const float* __restrict__ X, float* __restrict__ colsum, int L) {
    const int col = threadIdx.x;           // 0..127 (L == 128)
    const int r0 = blockIdx.x * 128;
    float local = 0.0f;
    for (int r = r0; r < r0 + 128; ++r) local += X[(size_t)r * L + col];
    atomicAdd(&colsum[col], local);
}

// ---------------------------------------------------------------------------
// Kernel 7: combine. num_neg == P deterministically (candidates >> P).
// ---------------------------------------------------------------------------
__global__ void __launch_bounds__(1)
combine(const float* __restrict__ acc, const float* __restrict__ colsum,
        float* __restrict__ out, int Bn, int P, int BL, int L) {
    float bb = 0.0f;
    for (int c = 0; c < L; ++c) {
        const float m = colsum[c] / (float)Bn;
        bb += m * m;
    }
    const float sim = (acc[0] + acc[1]) / (float)(2 * P);
    const float quant = acc[2] / (float)BL;
    out[0] = 1.0f * sim + 0.5f * quant + 0.1f * bb;
}

extern "C" void kernel_launch(void* const* d_in, const int* in_sizes, int n_in,
                              void* d_out, int out_size, void* d_ws, size_t ws_size,
                              hipStream_t stream) {
    const float*     bh  = (const float*)d_in[0];
    const float*     ch  = (const float*)d_in[1];
    const long long* pos = (const long long*)d_in[2];

    const int L  = 128;
    const int BL = in_sizes[0];
    const int Bn = BL / L;               // 8192
    const int P  = in_sizes[2] / 2;      // 4096
    const float margin = 0.5f * (float)L;

    // Workspace layout:
    //  acc[0]=pos_loss  acc[1]=neg_loss  acc[2]=quant_sum
    //  colsum[128] at acc+8 ; gram (16*Bn floats) at acc+256 ; mask bytes after.
    float* acc    = (float*)d_ws;
    float* colsum = acc + 8;
    float* gram   = acc + 256;
    unsigned char* mask = (unsigned char*)(gram + (size_t)GRAM_ROWS * Bn);

    hipMemsetAsync(d_ws, 0, 1024, stream);
    hipMemsetAsync(mask, 0, (size_t)GRAM_ROWS * Bn, stream);

    gram_slab_wmma<<<Bn / 16, 32, 0, stream>>>(bh, gram, Bn, L);
    pos_pairs<<<(P + 7) / 8, 256, 0, stream>>>(bh, pos, &acc[0], P, L, margin);
    mark_mask<<<(P + 255) / 256, 256, 0, stream>>>(pos, mask, Bn, P);
    neg_select<<<1, 1024, 0, stream>>>(gram, mask, &acc[1], Bn, P, margin);
    quant_sum<<<1024, 256, 0, stream>>>(ch, &acc[2], BL);
    bit_colsum<<<Bn / 128, 128, 0, stream>>>(bh, colsum, L);
    combine<<<1, 1, 0, stream>>>(acc, colsum, (float*)d_out, Bn, P, BL, L);
}
